// DeformableAttention_54958401519795
// MI455X (gfx1250) — matrix operand order repaired and checked by reading, not compiled, not run on hardware
//
#include <hip/hip_runtime.h>

// ---------------------------------------------------------------------------
// Deformable attention for MI455X (gfx1250, wave32, WMMA).
//   B=4, C=256, H=W=128, N=4.
// Phase 1: fused Q/K/V 1x1-conv GEMMs via v_wmma_f32_16x16x32_bf16
//          (bf16 inputs, fp32 accumulate), outputs stored PIXEL-MAJOR
//          (Qt/Kt/Vt[b][p][c]) so the gather phase reads contiguous columns.
// Phase 2: per-pixel offsets -> clamped indices -> gathered dot products.
//          Gathered K/V columns are 1KB contiguous and L2-resident
//          (per-batch K+V = 32MB << 192MB L2).
// Scratch: d_ws needs 3 * 4*16384*256 * 4B = 192 MB.
// ---------------------------------------------------------------------------

#define BATCH 4
#define CCH   256
#define HGT   128
#define WID   128
#define HWPIX (HGT * WID)   // 16384
#define NHEAD 4

typedef __attribute__((ext_vector_type(16))) __bf16         v16bf;
typedef __attribute__((ext_vector_type(8)))  __bf16         v8bf;
typedef __attribute__((ext_vector_type(8)))  float          v8f;
typedef __attribute__((ext_vector_type(4)))  unsigned short us4;

__device__ __forceinline__ unsigned short f2bf(float f) {
  unsigned u = __float_as_uint(f);
  u += 0x7FFFu + ((u >> 16) & 1u);   // round-to-nearest-even
  return (unsigned short)(u >> 16);
}

// ------------------------- Phase 1: QKV projections -------------------------
#define BN          128   // pixels per block tile
#define SX_STRIDE   264   // ushorts per sX row   (256 + pad; 528B, 16B aligned)
#define SW_STRIDE   40    // ushorts per sW row   (32 + pad;   80B, 16B aligned)
#define SOUT_STRIDE 260   // floats per staged output row

__global__ __launch_bounds__(256)
void qkv_proj_kernel(const float* __restrict__ x,
                     const float* __restrict__ Wq, const float* __restrict__ bq,
                     const float* __restrict__ Wk, const float* __restrict__ bk,
                     const float* __restrict__ Wv, const float* __restrict__ bv,
                     float* __restrict__ Qt, float* __restrict__ Kt,
                     float* __restrict__ Vt)
{
  extern __shared__ unsigned char smem[];
  unsigned short* sX   = (unsigned short*)smem;            // [BN][SX_STRIDE]
  unsigned short* sW   = sX + BN * SX_STRIDE;              // [256][SW_STRIDE]
  float*          sBias = (float*)(sW + 256 * SW_STRIDE);  // [256]
  float*          sOut  = sBias + 256;                     // [8][16][SOUT_STRIDE]

  const int t     = threadIdx.x;
  const int lane  = t & 31;
  const int wv    = t >> 5;
  const int l16   = lane & 15;
  const int khalf = lane >> 4;          // 0 or 1 (lane half, per ISA layouts)

  const int tile = blockIdx.x;          // 512 blocks = 4 batches * 128 tiles
  const int b    = tile >> 7;
  const int p0   = (tile & 127) * BN;

  // ---- load + transpose x tile into LDS as bf16: sX[pixel][channel] ----
  {
    const int pq = (t & 31) * 4;        // pixel quad (covers 128 pixels/wave)
    const int c0 = t >> 5;
    for (int pass = 0; pass < 32; ++pass) {
      const int c = c0 + 8 * pass;
      const float4 xv =
          *(const float4*)(x + ((size_t)(b * CCH + c)) * HWPIX + p0 + pq);
      sX[(pq + 0) * SX_STRIDE + c] = f2bf(xv.x);
      sX[(pq + 1) * SX_STRIDE + c] = f2bf(xv.y);
      sX[(pq + 2) * SX_STRIDE + c] = f2bf(xv.z);
      sX[(pq + 3) * SX_STRIDE + c] = f2bf(xv.w);
    }
  }

  const int n0 = wv * 16;               // wave's 16-pixel sub-column

  for (int which = 0; which < 3; ++which) {
    const float* Wsrc = (which == 0) ? Wq : (which == 1) ? Wk : Wv;
    const float* bsrc = (which == 0) ? bq : (which == 1) ? bk : bv;
    float*       dst  = (which == 0) ? Qt : (which == 1) ? Kt : Vt;

    __syncthreads();                    // sW/sBias reuse fence
    sBias[t] = bsrc[t];

    v8f acc[16] = {};                   // 16 M-tiles x (16x16 f32) = 128 VGPRs

    for (int ks = 0; ks < 8; ++ks) {    // K = 256 in steps of 32
      __syncthreads();
      {                                 // stream weight slab -> LDS (bf16)
        const float* wrow = Wsrc + (size_t)t * CCH + ks * 32;
        #pragma unroll
        for (int i = 0; i < 8; ++i) {
          const float4 wvx = *(const float4*)(wrow + 4 * i);
          us4 pk = { f2bf(wvx.x), f2bf(wvx.y), f2bf(wvx.z), f2bf(wvx.w) };
          *(us4*)(sW + t * SW_STRIDE + 4 * i) = pk;
        }
      }
      __syncthreads();

      // B fragment (x): lanes 0-15 hold K{0..7,16..23},
      // lanes 16-31 hold K{8..15,24..31} of its pixel column.
      const unsigned short* brow =
          sX + (n0 + l16) * SX_STRIDE + ks * 32 + khalf * 8;
      union { v16bf v; v8bf h[2]; } bf;
      bf.h[0] = *(const v8bf*)(brow);
      bf.h[1] = *(const v8bf*)(brow + 16);

      #pragma unroll
      for (int m = 0; m < 16; ++m) {
        // A fragment (weights): lanes 0-15 hold K{0..7,16..23},
        // lanes 16-31 hold K{8..15,24..31} of row M = m*16 + l16.
        const unsigned short* arow =
            sW + (m * 16 + l16) * SW_STRIDE + khalf * 8;
        union { v16bf v; v8bf h[2]; } af;
        af.h[0] = *(const v8bf*)(arow);
        af.h[1] = *(const v8bf*)(arow + 16);
        acc[m] = __builtin_amdgcn_wmma_f32_16x16x32_bf16(
            false, af.v, false, bf.v, (short)0, acc[m], false, false);
      }
    }

    // ---- stage accumulators (+bias) to LDS, then coalesced global store ----
    float* sOutW = sOut + wv * 16 * SOUT_STRIDE;
    #pragma unroll
    for (int m = 0; m < 16; ++m) {
      #pragma unroll
      for (int i = 0; i < 8; ++i) {
        const int o = m * 16 + i + 8 * khalf;  // C/D layout: M=i(+8), N=l16
        sOutW[l16 * SOUT_STRIDE + o] = acc[m][i] + sBias[o];
      }
    }
    for (int r = 0; r < 16; ++r) {      // 16 pixel rows, 1KB contiguous each
      float* drow = dst + ((size_t)(b * HWPIX + p0 + n0 + r)) * CCH;
      #pragma unroll
      for (int pass = 0; pass < 2; ++pass) {
        const int o4 = pass * 128 + lane * 4;
        *(float4*)(drow + o4) = *(const float4*)(sOutW + r * SOUT_STRIDE + o4);
      }
    }
  }
}

// ----------------- Phase 2: offsets, gather, attention, output --------------
__global__ __launch_bounds__(256)
void deform_attn_kernel(const float* __restrict__ Qt,
                        const float* __restrict__ Kt,
                        const float* __restrict__ Vt,
                        const float* __restrict__ Wo,
                        const float* __restrict__ bo,
                        float* __restrict__ out)
{
  __shared__ float sWo[8 * 256];
  __shared__ float sBo[8];
  __shared__ float sOut[8][256];

  const int t    = threadIdx.x;
  const int lane = t & 31;
  const int wv   = t >> 5;              // one wave32 per pixel

  const int gid = blockIdx.x;           // 8192 blocks = 4 batches * 2048 tiles
  const int b   = gid >> 11;
  const int p0  = (gid & 2047) * 8;

  #pragma unroll
  for (int j = 0; j < 8; ++j) sWo[j * 256 + t] = Wo[j * 256 + t];
  if (t < 8) sBo[t] = bo[t];
  __syncthreads();

  const int p  = p0 + wv;
  const int c0 = lane * 8;              // each lane owns 8 channels

  // Q column (contiguous, coalesced within the wave)
  float q[8];
  {
    const float* qcol = Qt + ((size_t)(b * HWPIX + p)) * CCH + c0;
    float4 q0 = *(const float4*)(qcol);
    float4 q1 = *(const float4*)(qcol + 4);
    q[0]=q0.x; q[1]=q0.y; q[2]=q0.z; q[3]=q0.w;
    q[4]=q1.x; q[5]=q1.y; q[6]=q1.z; q[7]=q1.w;
  }

  // offsets[j] = bo[j] + Wo[j,:] . Q[:,p]   (wave32 butterfly reduction)
  float offs[8];
  #pragma unroll
  for (int j = 0; j < 8; ++j) {
    const float* wrow = sWo + j * 256 + c0;
    float s = 0.f;
    #pragma unroll
    for (int i = 0; i < 8; ++i) s += wrow[i] * q[i];
    #pragma unroll
    for (int d = 16; d > 0; d >>= 1) s += __shfl_xor(s, d, 32);
    offs[j] = s + sBo[j];
  }

  // clamp then truncate toward zero (matches reference clip + astype(int32))
  const int ph = p >> 7, pw = p & 127;
  int idx[NHEAD];
  #pragma unroll
  for (int n = 0; n < NHEAD; ++n) {
    float fw = fminf(fmaxf((float)pw + offs[2 * n + 0], 0.f), (float)(WID - 1));
    float fh = fminf(fmaxf((float)ph + offs[2 * n + 1], 0.f), (float)(HGT - 1));
    idx[n] = ((int)fh) * WID + (int)fw;
  }

  // attn[n] = Q . K[:, idx[n]]   (gathered column is contiguous -> L2 hit)
  float attn[NHEAD];
  #pragma unroll
  for (int n = 0; n < NHEAD; ++n) {
    const float* kcol = Kt + ((size_t)(b * HWPIX + idx[n])) * CCH + c0;
    float4 k0 = *(const float4*)(kcol);
    float4 k1 = *(const float4*)(kcol + 4);
    float s = q[0]*k0.x + q[1]*k0.y + q[2]*k0.z + q[3]*k0.w
            + q[4]*k1.x + q[5]*k1.y + q[6]*k1.z + q[7]*k1.w;
    #pragma unroll
    for (int d = 16; d > 0; d >>= 1) s += __shfl_xor(s, d, 32);
    attn[n] = s;
  }

  // out[:,p] = sum_n attn[n] * V[:, idx[n]]
  float o8[8] = {0.f, 0.f, 0.f, 0.f, 0.f, 0.f, 0.f, 0.f};
  #pragma unroll
  for (int n = 0; n < NHEAD; ++n) {
    const float* vcol = Vt + ((size_t)(b * HWPIX + idx[n])) * CCH + c0;
    float4 v0 = *(const float4*)(vcol);
    float4 v1 = *(const float4*)(vcol + 4);
    o8[0] += attn[n] * v0.x; o8[1] += attn[n] * v0.y;
    o8[2] += attn[n] * v0.z; o8[3] += attn[n] * v0.w;
    o8[4] += attn[n] * v1.x; o8[5] += attn[n] * v1.y;
    o8[6] += attn[n] * v1.z; o8[7] += attn[n] * v1.w;
  }
  #pragma unroll
  for (int i = 0; i < 8; ++i) sOut[wv][c0 + i] = o8[i];
  __syncthreads();

  // transpose back to channel-major output, 32B-coalesced chunks
  const int lp = t & 7, cb = (t >> 3) * 8;
  #pragma unroll
  for (int i = 0; i < 8; ++i) {
    const int c = cb + i;
    out[((size_t)(b * CCH + c)) * HWPIX + p0 + lp] = sOut[lp][c];
  }
}

// ---------------------------------------------------------------------------
extern "C" void kernel_launch(void* const* d_in, const int* in_sizes, int n_in,
                              void* d_out, int out_size, void* d_ws,
                              size_t ws_size, hipStream_t stream) {
  (void)in_sizes; (void)n_in; (void)out_size; (void)ws_size;

  const float* x  = (const float*)d_in[0];
  const float* Wq = (const float*)d_in[1];
  const float* bq = (const float*)d_in[2];
  const float* Wk = (const float*)d_in[3];
  const float* bk = (const float*)d_in[4];
  const float* Wv = (const float*)d_in[5];
  const float* bv = (const float*)d_in[6];
  const float* Wo = (const float*)d_in[7];
  const float* bo = (const float*)d_in[8];
  float* out = (float*)d_out;

  const size_t planeElems = (size_t)BATCH * HWPIX * CCH;  // 16,777,216
  float* Qt = (float*)d_ws;
  float* Kt = Qt + planeElems;
  float* Vt = Kt + planeElems;

  const int SMEM_A = BN * SX_STRIDE * 2     // x tile (bf16)
                   + 256 * SW_STRIDE * 2    // weight slab (bf16)
                   + 256 * 4                // bias
                   + 8 * 16 * SOUT_STRIDE * 4;  // store staging  (~218KB < 320KB WGP LDS)
  hipFuncSetAttribute((const void*)qkv_proj_kernel,
                      hipFuncAttributeMaxDynamicSharedMemorySize, SMEM_A);

  qkv_proj_kernel<<<dim3(512), dim3(256), SMEM_A, stream>>>(
      x, Wq, bq, Wk, bk, Wv, bv, Qt, Kt, Vt);
  deform_attn_kernel<<<dim3(8192), dim3(256), 0, stream>>>(
      Qt, Kt, Vt, Wo, bo, out);
}
